// LapCluster_1838246002783
// MI455X (gfx1250) — matrix operand order repaired and checked by reading, not compiled, not run on hardware
//
#include <hip/hip_runtime.h>
#include <cstdint>
#include <cstddef>

typedef __bf16 v16bf __attribute__((ext_vector_type(16)));
typedef __bf16 v8bf  __attribute__((ext_vector_type(8)));
typedef float  v8f   __attribute__((ext_vector_type(8)));

#define B_   8
#define N_   32768
#define C_   32
#define CAT_ 40
#define SA   136   // LDS stride (bf16 elems) for 128-ch buffer (272B, 16B-multiple, bank-skewed)
#define SB   264   // LDS stride for 256-ch buffer (528B)

// bf16 weight offsets (elements) inside workspace
constexpr int OFF_WIN = 0;        // 256x32 (padded from 256x22)
constexpr int OFF_W1  = 8192;     // 2 x 64x256
constexpr int OFF_W2  = 40960;    // 2 x 128x64
constexpr int OFF_W3  = 57344;    // 2 x 128x128
constexpr int OFF_WC1 = 90112;    // 2 x 128x128
constexpr int OFF_WC2 = 122880;   // 2 x 256x128
constexpr int OFF_WC3 = 188416;   // 2 x 32x256
constexpr int OFF_WO1 = 204800;   // 128x256
constexpr int OFF_WO2 = 237568;   // 128x128

constexpr size_t WS_FBUF_BYTE = 524288;     // bf16 [B][N][128]  (64 MB, fits in 192MB L2)
constexpr size_t WS_FLT_BYTE  = 67633152;   // float region
constexpr int F_SEGF0 = 0;        // B*C*128
constexpr int F_SEGF1 = 32768;
constexpr int F_SEGC0 = 65536;    // B*C*32
constexpr int F_SEGC1 = 73728;
constexpr int F_CF0   = 81920;    // B*C*128 pooled
constexpr int F_CF1   = 114688;
constexpr int F_NETM  = 147456;   // B*128

// -------- CDNA5 async global->LDS path (guarded; falls back to vector copy) --------
#if defined(__HIP_DEVICE_COMPILE__) && __has_builtin(__builtin_amdgcn_global_load_async_to_lds_b128)
#define ASYNC_LDS 1
typedef int vsi4 __attribute__((vector_size(16)));                  // b128 payload type
typedef __attribute__((address_space(1))) vsi4* g_vsi4p;            // global (prints as __device__)
typedef __attribute__((address_space(3))) vsi4* l_vsi4p;            // LDS
#endif

__device__ __forceinline__ void async_wait0() {
#if defined(__HIP_DEVICE_COMPILE__) && __has_builtin(__builtin_amdgcn_s_wait_asynccnt)
  __builtin_amdgcn_s_wait_asynccnt(0);
#else
  asm volatile("s_wait_asynccnt 0x0" ::: "memory");
#endif
}

// Copy one 16-point x 128-ch bf16 tile (4KB) from global [pt*128+ch] into LDS [pt*SA+ch].
__device__ __forceinline__ void prefetch_tile(const __bf16* __restrict__ src,
                                              __bf16* dst, int lane) {
#ifdef ASYNC_LDS
  for (int idx = lane; idx < 256; idx += 32) {          // 256 x 16B chunks
    int pt = idx >> 4, co = (idx & 15) * 8;
    __builtin_amdgcn_global_load_async_to_lds_b128(
        (g_vsi4p)(src + (size_t)pt * 128 + co),
        (l_vsi4p)(unsigned)(size_t)(dst + pt * SA + co),   // low 32 bits = LDS byte offset
        0, 0);
  }
#else
  for (int idx = lane; idx < 256; idx += 32) {
    int pt = idx >> 4, co = (idx & 15) * 8;
    *(v8bf*)(dst + pt * SA + co) = *(const v8bf*)(src + (size_t)pt * 128 + co);
  }
#endif
}

// ---------------- helpers ----------------
__device__ __forceinline__ void atomicMaxF(float* addr, float v) {
  if (v >= 0.0f) atomicMax((int*)addr, __float_as_int(v));
  else           atomicMin((unsigned int*)addr, __float_as_uint(v));
}

// One conv layer on a 16-point tile: out[16ch-tile][16pt] = relu(W x Act + bias)
// LDS activation layout: [pt][ch] with stride SIN/SOUT.
template<int IN_CH, int OUT_CH, int SIN, int SOUT>
__device__ __forceinline__ void layer_wmma(const __bf16* __restrict__ W,
                                           const float* bias,
                                           const __bf16* bin, __bf16* bout, int lane)
{
  const int col = lane & 15;   // point / A-row index
  const int hf  = lane >> 4;   // wave half
  for (int mb = 0; mb < OUT_CH; mb += 16) {
    v8f acc = {};
    const __bf16* wrow = W + (size_t)(mb + col) * IN_CH;
    #pragma unroll
    for (int kb = 0; kb < IN_CH; kb += 32) {
      // B fragment: element j -> K = kb + hf*16 + j  (contiguous in [pt][ch] layout)
      v8bf b0 = *(const v8bf*)(bin + col * SIN + kb + hf * 16);
      v8bf b1 = *(const v8bf*)(bin + col * SIN + kb + hf * 16 + 8);
      v16bf bfr = __builtin_shufflevector(b0, b1, 0,1,2,3,4,5,6,7,8,9,10,11,12,13,14,15);
      // A fragment: element j -> K = kb + hf*16 + j (matches B)
      v8bf a0 = *(const v8bf*)(wrow + kb + hf * 16);
      v8bf a1 = *(const v8bf*)(wrow + kb + hf * 16 + 8);
      v16bf afr = __builtin_shufflevector(a0, a1, 0,1,2,3,4,5,6,7,8,9,10,11,12,13,14,15);
      acc = __builtin_amdgcn_wmma_f32_16x16x32_bf16(false, afr, false, bfr,
                                                    (short)0, acc, false, false);
    }
    const int mrow = mb + hf * 8;  // D: VGPR r -> row mb + r + 8*hf (contiguous per lane)
    #pragma unroll
    for (int r = 0; r < 8; ++r) {
      float v = acc[r] + bias[mrow + r];
      v = fmaxf(v, 0.0f);
      bout[col * SOUT + mrow + r] = (__bf16)v;
    }
  }
}

// ---------------- setup kernels ----------------
__global__ void k_cvt(const float* __restrict__ s, __bf16* __restrict__ d, int n) {
  int i = blockIdx.x * blockDim.x + threadIdx.x;
  if (i < n) d[i] = (__bf16)s[i];
}
__global__ void k_cvt_win(const float* __restrict__ s, __bf16* __restrict__ d) {
  int i = blockIdx.x * blockDim.x + threadIdx.x;  // 256x32, pad cols 22..31
  if (i < 256 * 32) { int r = i >> 5, c = i & 31; d[i] = (__bf16)((c < 22) ? s[r * 22 + c] : 0.0f); }
}
__global__ void k_init(float* __restrict__ f) {
  int i = blockIdx.x * blockDim.x + threadIdx.x;
  if (i < 81920) f[i] = -100.0f;                // segf0/1, segc0/1 (= the -100 clamp)
  if (i < 1024)  f[F_NETM + i] = 0.0f;          // global max over ReLU output
}

// ---------------- phase 1: input -> block0 f2 + corr segmax ----------------
__global__ __launch_bounds__(64)
void k_block0(const float* __restrict__ feat, const int* __restrict__ clus0,
              const __bf16* __restrict__ wbf, const float* __restrict__ b_in,
              const float* __restrict__ b1, const float* __restrict__ b2,
              const float* __restrict__ bc1, const float* __restrict__ bc2,
              const float* __restrict__ bc3,
              __bf16* __restrict__ fbuf, float* __restrict__ gF, float* __restrict__ gC)
{
  __shared__ __align__(16) __bf16 A[2][16 * SA];
  __shared__ __align__(16) __bf16 Bm[2][16 * SB];
  __shared__ float segF[C_ * 128];
  __shared__ float segC[C_ * 32];
  __shared__ float bS[864];   // b_in@0, b1@256, b2@320, bc1@448, bc2@576, bc3@832
  __shared__ int cid[2][16];
  const int b = blockIdx.x >> 7, nb = blockIdx.x & 127;
  const int wave = threadIdx.x >> 5, lane = threadIdx.x & 31, tid = threadIdx.x;
  for (int i = tid; i < 256; i += 64) bS[i]       = b_in[i];
  for (int i = tid; i < 64;  i += 64) bS[256 + i] = b1[i];
  for (int i = tid; i < 128; i += 64) bS[320 + i] = b2[i];
  for (int i = tid; i < 128; i += 64) bS[448 + i] = bc1[i];
  for (int i = tid; i < 256; i += 64) bS[576 + i] = bc2[i];
  for (int i = tid; i < 32;  i += 64) bS[832 + i] = bc3[i];
  for (int i = tid; i < C_ * 128; i += 64) segF[i] = -1e30f;
  for (int i = tid; i < C_ * 32;  i += 64) segC[i] = -1e30f;
  __syncthreads();
  __bf16* a = A[wave]; __bf16* bb = Bm[wave];
  const __bf16 *Wi = wbf + OFF_WIN, *W1 = wbf + OFF_W1, *W2 = wbf + OFF_W2;
  const __bf16 *Wc1 = wbf + OFF_WC1, *Wc2 = wbf + OFF_WC2, *Wc3 = wbf + OFF_WC3;
  for (int t = 0; t < 8; ++t) {
    const int n0 = nb * 256 + wave * 128 + t * 16;
    for (int i = lane; i < 16 * 32; i += 32) {        // features (22ch, zero-pad to 32)
      int pt = i >> 5, ch = i & 31;
      float v = (ch < 22) ? feat[(size_t)(b * N_ + n0 + pt) * 22 + ch] : 0.0f;
      a[pt * SA + ch] = (__bf16)v;
    }
    if (lane < 16) cid[wave][lane] = clus0[(size_t)b * N_ + n0 + lane];
    __syncthreads();
    layer_wmma<32, 256, SA, SB>(Wi, bS, a, bb, lane);        // f0 (256) -> B
    __syncthreads();
    layer_wmma<256, 64, SB, SA>(W1, bS + 256, bb, a, lane);  // f1 (64)  -> A
    __syncthreads();
    layer_wmma<64, 128, SA, SB>(W2, bS + 320, a, bb, lane);  // f2 (128) -> B
    __syncthreads();
    for (int idx = lane; idx < 256; idx += 32) {             // store f2 (b128 chunks)
      int pt = idx >> 4, co = (idx & 15) * 8;
      *(v8bf*)(fbuf + (size_t)(b * N_ + n0 + pt) * 128 + co) = *(const v8bf*)(bb + pt * SB + co);
    }
    for (int i = lane; i < 16 * 128; i += 32) {              // segmax f2
      int pt = i >> 7, ch = i & 127;
      atomicMaxF(&segF[cid[wave][pt] * 128 + ch], (float)bb[pt * SB + ch]);
    }
    __syncthreads();
    layer_wmma<128, 128, SB, SA>(Wc1, bS + 448, bb, a, lane);  // c1 (128) -> A
    __syncthreads();
    layer_wmma<128, 256, SA, SB>(Wc2, bS + 576, a, bb, lane);  // c2 (256) -> B
    __syncthreads();
    layer_wmma<256, 32, SB, SA>(Wc3, bS + 832, bb, a, lane);   // c3 (32)  -> A
    __syncthreads();
    for (int i = lane; i < 16 * 32; i += 32) {                 // segmax c3
      int pt = i >> 5, ch = i & 31;
      atomicMaxF(&segC[cid[wave][pt] * 32 + ch], (float)a[pt * SA + ch]);
    }
    __syncthreads();
  }
  __syncthreads();
  for (int i = tid; i < C_ * 128; i += 64) atomicMaxF(&gF[(size_t)b * 4096 + i], segF[i]);
  for (int i = tid; i < C_ * 32;  i += 64) atomicMaxF(&gC[(size_t)b * 1024 + i], segC[i]);
}

// ---------------- pooling math (per batch, tiny) ----------------
__global__ __launch_bounds__(128)
void k_pool(const float* __restrict__ gF, const float* __restrict__ gC, float* __restrict__ cf)
{
  const int b = blockIdx.x, tid = threadIdx.x;
  __shared__ float cm[C_ * 32], corr[C_ * C_], sF[C_ * 128], rs[C_];
  for (int i = tid; i < C_ * 32;  i += 128) cm[i] = gC[(size_t)b * 1024 + i];
  for (int i = tid; i < C_ * 128; i += 128) sF[i] = gF[(size_t)b * 4096 + i];
  __syncthreads();
  if (tid < C_) {
    float s = 0.f;
    for (int j = 0; j < 32; ++j) { float x = cm[tid * 32 + j]; s += x * x; }
    rs[tid] = 1.0f / fmaxf(sqrtf(s), 1e-12f);
  }
  __syncthreads();
  for (int i = tid; i < C_ * C_; i += 128) {
    int c1 = i >> 5, c2 = i & 31; float d = 0.f;
    for (int j = 0; j < 32; ++j) d += cm[c1 * 32 + j] * cm[c2 * 32 + j];
    corr[i] = d * rs[c1] * rs[c2];
  }
  __syncthreads();
  for (int i = tid; i < C_ * 128; i += 128) {
    int c = i >> 7, ch = i & 127; float s = 0.f;
    for (int c2 = 0; c2 < C_; ++c2) s += sF[c2 * 128 + ch] * corr[c2 * 32 + c];
    cf[(size_t)b * 4096 + c * 128 + ch] = s;
  }
}

// ---------------- phase 2: f2 -> block1 f + corr segmax ----------------
__global__ __launch_bounds__(64)
void k_block1(const int* __restrict__ clus0, const int* __restrict__ clus1,
              const __bf16* __restrict__ wbf,
              const float* __restrict__ b3_0, const float* __restrict__ b1_1,
              const float* __restrict__ b2_1, const float* __restrict__ bc1_1,
              const float* __restrict__ bc2_1, const float* __restrict__ bc3_1,
              const float* __restrict__ cf0, __bf16* __restrict__ fbuf,
              float* __restrict__ gF, float* __restrict__ gC)
{
  __shared__ __align__(16) __bf16 A[2][2][16 * SA];   // [buf][wave] double-buffered
  __shared__ __align__(16) __bf16 Bm[2][16 * SB];
  __shared__ float segF[C_ * 128];
  __shared__ float segC[C_ * 32];
  __shared__ float bS[736];  // b3_0@0, b1_1@128, b2_1@192, bc1_1@320, bc2_1@448, bc3_1@704
  __shared__ int c0[2][16], c1a[2][16];
  const int b = blockIdx.x >> 7, nb = blockIdx.x & 127;
  const int wave = threadIdx.x >> 5, lane = threadIdx.x & 31, tid = threadIdx.x;
  for (int i = tid; i < 128; i += 64) bS[i]       = b3_0[i];
  for (int i = tid; i < 64;  i += 64) bS[128 + i] = b1_1[i];
  for (int i = tid; i < 128; i += 64) bS[192 + i] = b2_1[i];
  for (int i = tid; i < 128; i += 64) bS[320 + i] = bc1_1[i];
  for (int i = tid; i < 256; i += 64) bS[448 + i] = bc2_1[i];
  for (int i = tid; i < 32;  i += 64) bS[704 + i] = bc3_1[i];
  for (int i = tid; i < C_ * 128; i += 64) segF[i] = -1e30f;
  for (int i = tid; i < C_ * 32;  i += 64) segC[i] = -1e30f;
  __bf16* bb = Bm[wave];
  const __bf16 *W3 = wbf + OFF_W3, *W1 = wbf + OFF_W1 + 16384, *W2 = wbf + OFF_W2 + 8192;
  const __bf16 *Wc1 = wbf + OFF_WC1 + 16384, *Wc2 = wbf + OFF_WC2 + 32768, *Wc3 = wbf + OFF_WC3 + 8192;
  const size_t pbase = (size_t)(b * N_ + nb * 256 + wave * 128) * 128;
  prefetch_tile(fbuf + pbase, A[0][wave], lane);             // tile 0
  for (int t = 0; t < 8; ++t) {
    const int n0 = nb * 256 + wave * 128 + t * 16;
    __bf16* cur = A[t & 1][wave];
    __bf16* nxt = A[(t + 1) & 1][wave];
    if (lane < 16) {
      c0[wave][lane]  = clus0[(size_t)b * N_ + n0 + lane];
      c1a[wave][lane] = clus1[(size_t)b * N_ + n0 + lane];
    }
    async_wait0();
    __syncthreads();
    if (t < 7) prefetch_tile(fbuf + pbase + (size_t)(t + 1) * 16 * 128, nxt, lane);
    layer_wmma<128, 128, SA, SB>(W3, bS, cur, bb, lane);     // f3 -> B rows 0..127
    for (int idx = lane; idx < 256; idx += 32) {             // pool gather -> B rows 128..255
      int pt = idx >> 4, co = (idx & 15) * 8;
      const float* s = cf0 + ((size_t)b * C_ + c0[wave][pt]) * 128 + co;
      v8bf tv;
      #pragma unroll
      for (int j = 0; j < 8; ++j) tv[j] = (__bf16)s[j];
      *(v8bf*)(bb + pt * SB + 128 + co) = tv;
    }
    __syncthreads();
    layer_wmma<256, 64, SB, SA>(W1, bS + 128, bb, cur, lane);
    __syncthreads();
    layer_wmma<64, 128, SA, SB>(W2, bS + 192, cur, bb, lane);  // f_b1 -> B
    __syncthreads();
    for (int idx = lane; idx < 256; idx += 32) {               // store f_b1 (in place, b128)
      int pt = idx >> 4, co = (idx & 15) * 8;
      *(v8bf*)(fbuf + (size_t)(b * N_ + n0 + pt) * 128 + co) = *(const v8bf*)(bb + pt * SB + co);
    }
    for (int i = lane; i < 16 * 128; i += 32) {                // segmax f_b1
      int pt = i >> 7, ch = i & 127;
      atomicMaxF(&segF[c1a[wave][pt] * 128 + ch], (float)bb[pt * SB + ch]);
    }
    __syncthreads();
    layer_wmma<128, 128, SB, SA>(Wc1, bS + 320, bb, cur, lane);
    __syncthreads();
    layer_wmma<128, 256, SA, SB>(Wc2, bS + 448, cur, bb, lane);
    __syncthreads();
    layer_wmma<256, 32, SB, SA>(Wc3, bS + 704, bb, cur, lane);
    __syncthreads();
    for (int i = lane; i < 16 * 32; i += 32) {
      int pt = i >> 5, ch = i & 31;
      atomicMaxF(&segC[c1a[wave][pt] * 32 + ch], (float)cur[pt * SA + ch]);
    }
    __syncthreads();
  }
  __syncthreads();
  for (int i = tid; i < C_ * 128; i += 64) atomicMaxF(&gF[(size_t)b * 4096 + i], segF[i]);
  for (int i = tid; i < C_ * 32;  i += 64) atomicMaxF(&gC[(size_t)b * 1024 + i], segC[i]);
}

// ---------------- phase 3: out convs + global max ----------------
__global__ __launch_bounds__(64)
void k_final(const int* __restrict__ clus1, const __bf16* __restrict__ wbf,
             const float* __restrict__ b3_1, const float* __restrict__ bo1,
             const float* __restrict__ bo2, const float* __restrict__ cf1,
             const __bf16* __restrict__ fbuf, float* __restrict__ gnet)
{
  __shared__ __align__(16) __bf16 A[2][2][16 * SA];
  __shared__ __align__(16) __bf16 Bm[2][16 * SB];
  __shared__ float nmax[128];
  __shared__ float bS[384];   // b3_1@0, bo1@128, bo2@256
  __shared__ int c1a[2][16];
  const int b = blockIdx.x >> 7, nb = blockIdx.x & 127;
  const int wave = threadIdx.x >> 5, lane = threadIdx.x & 31, tid = threadIdx.x;
  for (int i = tid; i < 128; i += 64) { bS[i] = b3_1[i]; bS[128 + i] = bo1[i]; bS[256 + i] = bo2[i]; }
  for (int i = tid; i < 128; i += 64) nmax[i] = -1e30f;
  __bf16* bb = Bm[wave];
  const __bf16 *W3 = wbf + OFF_W3 + 16384, *Wo1 = wbf + OFF_WO1, *Wo2 = wbf + OFF_WO2;
  const size_t pbase = (size_t)(b * N_ + nb * 256 + wave * 128) * 128;
  prefetch_tile(fbuf + pbase, A[0][wave], lane);
  for (int t = 0; t < 8; ++t) {
    const int n0 = nb * 256 + wave * 128 + t * 16;
    __bf16* cur = A[t & 1][wave];
    __bf16* nxt = A[(t + 1) & 1][wave];
    if (lane < 16) c1a[wave][lane] = clus1[(size_t)b * N_ + n0 + lane];
    async_wait0();
    __syncthreads();
    if (t < 7) prefetch_tile(fbuf + pbase + (size_t)(t + 1) * 16 * 128, nxt, lane);
    layer_wmma<128, 128, SA, SB>(W3, bS, cur, bb, lane);     // f4 -> B rows 0..127
    for (int idx = lane; idx < 256; idx += 32) {             // pool gather -> B rows 128..255
      int pt = idx >> 4, co = (idx & 15) * 8;
      const float* s = cf1 + ((size_t)b * C_ + c1a[wave][pt]) * 128 + co;
      v8bf tv;
      #pragma unroll
      for (int j = 0; j < 8; ++j) tv[j] = (__bf16)s[j];
      *(v8bf*)(bb + pt * SB + 128 + co) = tv;
    }
    __syncthreads();
    layer_wmma<256, 128, SB, SA>(Wo1, bS + 128, bb, cur, lane);
    __syncthreads();
    layer_wmma<128, 128, SA, SB>(Wo2, bS + 256, cur, bb, lane);  // o2 -> B
    __syncthreads();
    for (int i = lane; i < 16 * 128; i += 32) {
      int ch = i & 127;
      atomicMaxF(&nmax[ch], (float)bb[(i >> 7) * SB + ch]);
    }
    __syncthreads();
  }
  __syncthreads();
  for (int i = tid; i < 128; i += 64) atomicMaxF(&gnet[(size_t)b * 128 + i], nmax[i]);
}

// ---------------- dense head ----------------
__global__ __launch_bounds__(256)
void k_head(const float* __restrict__ gnet,
            const float* __restrict__ dw1, const float* __restrict__ db1,
            const float* __restrict__ dw2, const float* __restrict__ db2,
            const float* __restrict__ dw3, const float* __restrict__ db3,
            float* __restrict__ out)
{
  const int b = blockIdx.x, tid = threadIdx.x;
  __shared__ float nS[128], h1[256], h2[256];
  if (tid < 128) nS[tid] = gnet[(size_t)b * 128 + tid];
  __syncthreads();
  {
    float s = db1[tid];
    for (int k = 0; k < 128; ++k) s += nS[k] * dw1[k * 256 + tid];
    h1[tid] = (s >= 0.f) ? s : 0.2f * s;
  }
  __syncthreads();
  {
    float s = db2[tid];
    for (int k = 0; k < 256; ++k) s += h1[k] * dw2[k * 256 + tid];
    h2[tid] = (s >= 0.f) ? s : 0.2f * s;
  }
  __syncthreads();
  if (tid < CAT_) {
    float s = db3[tid];
    for (int k = 0; k < 256; ++k) s += h2[k] * dw3[k * CAT_ + tid];
    out[b * CAT_ + tid] = s;
  }
}

// ---------------- launch ----------------
extern "C" void kernel_launch(void* const* d_in, const int* in_sizes, int n_in,
                              void* d_out, int out_size, void* d_ws, size_t ws_size,
                              hipStream_t stream)
{
  const float* feat = (const float*)d_in[0];
  const int*   clus = (const int*)d_in[1];
  const float* w_in = (const float*)d_in[2];  const float* b_in = (const float*)d_in[3];
  const float* w1   = (const float*)d_in[4];  const float* b1   = (const float*)d_in[5];
  const float* w2   = (const float*)d_in[6];  const float* b2   = (const float*)d_in[7];
  const float* w3   = (const float*)d_in[8];  const float* b3   = (const float*)d_in[9];
  const float* wc1  = (const float*)d_in[10]; const float* bc1  = (const float*)d_in[11];
  const float* wc2  = (const float*)d_in[12]; const float* bc2  = (const float*)d_in[13];
  const float* wc3  = (const float*)d_in[14]; const float* bc3  = (const float*)d_in[15];
  const float* wo1  = (const float*)d_in[16]; const float* bo1  = (const float*)d_in[17];
  const float* wo2  = (const float*)d_in[18]; const float* bo2  = (const float*)d_in[19];
  const float* dw1  = (const float*)d_in[20]; const float* db1  = (const float*)d_in[21];
  const float* dw2  = (const float*)d_in[22]; const float* db2  = (const float*)d_in[23];
  const float* dw3  = (const float*)d_in[24]; const float* db3  = (const float*)d_in[25];
  float* out = (float*)d_out;

  char* ws = (char*)d_ws;
  __bf16* wbf  = (__bf16*)ws;
  __bf16* fbuf = (__bf16*)(ws + WS_FBUF_BYTE);
  float*  fws  = (float*)(ws + WS_FLT_BYTE);

  // weights -> bf16 (w_in zero-padded to K=32)
  k_cvt_win<<<32, 256, 0, stream>>>(w_in, wbf + OFF_WIN);
  k_cvt<<<128, 256, 0, stream>>>(w1,  wbf + OFF_W1,  32768);
  k_cvt<<<64,  256, 0, stream>>>(w2,  wbf + OFF_W2,  16384);
  k_cvt<<<128, 256, 0, stream>>>(w3,  wbf + OFF_W3,  32768);
  k_cvt<<<128, 256, 0, stream>>>(wc1, wbf + OFF_WC1, 32768);
  k_cvt<<<256, 256, 0, stream>>>(wc2, wbf + OFF_WC2, 65536);
  k_cvt<<<64,  256, 0, stream>>>(wc3, wbf + OFF_WC3, 16384);
  k_cvt<<<128, 256, 0, stream>>>(wo1, wbf + OFF_WO1, 32768);
  k_cvt<<<64,  256, 0, stream>>>(wo2, wbf + OFF_WO2, 16384);

  k_init<<<320, 256, 0, stream>>>(fws);

  k_block0<<<1024, 64, 0, stream>>>(feat, clus, wbf, b_in, b1, b2, bc1, bc2, bc3,
                                    fbuf, fws + F_SEGF0, fws + F_SEGC0);
  k_pool<<<8, 128, 0, stream>>>(fws + F_SEGF0, fws + F_SEGC0, fws + F_CF0);
  k_block1<<<1024, 64, 0, stream>>>(clus, clus + B_ * N_, wbf, b3, b1 + 64, b2 + 128,
                                    bc1 + 128, bc2 + 256, bc3 + 32,
                                    fws + F_CF0, fbuf, fws + F_SEGF1, fws + F_SEGC1);
  k_pool<<<8, 128, 0, stream>>>(fws + F_SEGF1, fws + F_SEGC1, fws + F_CF1);
  k_final<<<1024, 64, 0, stream>>>(clus + B_ * N_, wbf, b3 + 128, bo1, bo2,
                                   fws + F_CF1, fbuf, fws + F_NETM);
  k_head<<<8, 256, 0, stream>>>(fws + F_NETM, dw1, db1, dw2, db2, dw3, db3, out);
}
